// FineGrainedCellLSTM_63170378989793
// MI455X (gfx1250) — matrix operand order repaired and checked by reading, not compiled, not run on hardware
//
#include <hip/hip_runtime.h>
#include <hip/hip_bf16.h>

// ---------------------------------------------------------------------------
// FineGrainedCellLSTM on MI455X (gfx1250, wave32, WMMA + TDM)
//
// Phase A (parallel): xin = x @ Wi + bi  -> bf16, via v_wmma_f32_16x16x32_bf16
// Phase B (serial):   persistent kernel, 1024 phases (512 steps x 2 layers),
//                     grid barrier between phases. Per phase each WG computes
//                     gates[64 x 16cols x 4gates] = inp@Wxg + h@Whg (+biases).
//                     The A operand (activations) is staged into LDS by the
//                     Tensor Data Mover (tensor_load_to_lds, TENSORcnt),
//                     double-buffered so DMA overlaps WMMA compute; weights
//                     stream from L2 as pre-swizzled per-lane b128 fragments.
// ---------------------------------------------------------------------------

typedef __bf16 bf16_t;
typedef __attribute__((ext_vector_type(16))) __bf16 v16bf;
typedef __attribute__((ext_vector_type(8)))  __bf16 v8bf;
typedef __attribute__((ext_vector_type(8)))  float  v8f;
typedef __attribute__((ext_vector_type(4)))  unsigned int u32x4;
typedef __attribute__((ext_vector_type(8)))  int i32x8;
typedef __attribute__((ext_vector_type(4)))  int i32x4;

union BFrag { v16bf v; v8bf h[2]; };

#define SEQ 512
#define BATCH 64
#define INP 512
#define HID 1024
#define G4 4096   // 4*HID
#define NLAYER 2
#define ACHUNK 256  // A-tile chunk width (K columns) staged per TDM transfer

__device__ __forceinline__ v8f wmma_bf16(v16bf a, v16bf b, v8f c) {
  // 8-arg form: (neg_a, A, neg_b, B, c_mod, C, reuse_a, reuse_b)
  return __builtin_amdgcn_wmma_f32_16x16x32_bf16(false, a, false, b,
                                                 (short)0, c, false, false);
}

// A fragment (16x32 bf16, row-major source, lda elems).
// ISA layout: lanes 0-15: M=lane, K = {k0+0..7, k0+16..23}
//             lanes16-31: M=lane-16, K = {k0+8..15, k0+24..31}
__device__ __forceinline__ BFrag load_a_frag(const bf16_t* A, int lda,
                                             int row0, int k0, int lane) {
  const int half = lane >> 4;
  const int r = lane & 15;
  const bf16_t* p = A + (size_t)(row0 + r) * lda + k0 + half * 8;
  BFrag f;
  f.h[0] = *(const v8bf*)(p);
  f.h[1] = *(const v8bf*)(p + 16);
  return f;
}

// B fragment from pre-swizzled weights: [ntile][ktile][lane][16] contiguous.
__device__ __forceinline__ BFrag load_b_frag(const bf16_t* W, int ntile,
                                             int kt, int nkt, int lane) {
  const bf16_t* p = W + (((size_t)ntile * nkt + kt) * 32 + lane) * 16;
  BFrag f;
  f.h[0] = *(const v8bf*)(p);
  f.h[1] = *(const v8bf*)(p + 8);
  return f;
}

__device__ __forceinline__ float sigm(float x) {
  return 1.0f / (1.0f + __expf(-x));
}

// ---- Tensor Data Mover: DMA a [64 rows x ACHUNK cols] bf16 tile from a
// row-major [64 x 1024] global matrix into LDS (row-major [64][ACHUNK]).
// D# bit layout per CDNA5 ISA (groups 0/1; 2D tile).
__device__ __forceinline__ void tdm_load_a_tile(unsigned lds_off,
                                                const bf16_t* gptr) {
  unsigned long long ga = (unsigned long long)gptr;
  u32x4 g0;
  g0[0] = 1u;                                    // count=1, user descriptor
  g0[1] = lds_off;                               // lds_addr (bytes)
  g0[2] = (unsigned)(ga & 0xFFFFFFFFu);          // global_addr[31:0]
  g0[3] = (unsigned)((ga >> 32) & 0x01FFFFFFu)   // global_addr[56:32]
        | (2u << 30);                            // type=2 ("image")
  i32x8 g1;
  g1[0] = (int)(1u << 16);                       // data_size=1 (2 bytes)
  g1[1] = (int)((unsigned)HID << 16);            // tensor_dim0[15:0] = 1024
  g1[2] = (int)((unsigned)BATCH << 16);          // tensor_dim1[15:0] = 64
  g1[3] = (int)((unsigned)ACHUNK << 16);         // tile_dim0 = 256
  g1[4] = BATCH;                                 // tile_dim1 = 64
  g1[5] = HID;                                   // tensor_dim0_stride = 1024
  g1[6] = 0;
  g1[7] = 0;
  i32x4 z4 = {0, 0, 0, 0};
#if defined(__clang_major__) && __clang_major__ >= 23
  i32x8 z8 = {0, 0, 0, 0, 0, 0, 0, 0};
  __builtin_amdgcn_tensor_load_to_lds(g0, g1, z4, z4, z8, 0);
#else
  __builtin_amdgcn_tensor_load_to_lds(g0, g1, z4, z4, 0);
#endif
}

// --------------------------- prep kernels ----------------------------------

__global__ void cvt_f32_bf16_kernel(const float* __restrict__ src,
                                    bf16_t* __restrict__ dst, int n) {
  for (int i = blockIdx.x * blockDim.x + threadIdx.x; i < n;
       i += gridDim.x * blockDim.x)
    dst[i] = (bf16_t)src[i];
}

// src: [K][N] f32 row-major -> dst: WMMA-B fragment order, bf16.
__global__ void swizzle_w_kernel(const float* __restrict__ src,
                                 bf16_t* __restrict__ dst, int K, int N) {
  int total = K * N;
  int nkt = K >> 5;
  for (int i = blockIdx.x * blockDim.x + threadIdx.x; i < total;
       i += gridDim.x * blockDim.x) {
    int k = i / N, n = i - k * N;
    int lane = (n & 15) | (((k >> 3) & 1) << 4);
    size_t d = ((((size_t)(n >> 4) * nkt) + (k >> 5)) * 32 + lane) * 16 + ((k & 7) | (((k >> 4) & 1) << 3));
    dst[d] = (bf16_t)src[i];
  }
}

// --------------------------- xin GEMM --------------------------------------
__global__ void gemm_xin_kernel(const bf16_t* __restrict__ A,
                                const bf16_t* __restrict__ W,
                                const float* __restrict__ bias,
                                bf16_t* __restrict__ out) {
  const int mblk = blockIdx.x & 511;
  const int nblk = blockIdx.x >> 9;
  const int w = threadIdx.x >> 5;
  const int lane = threadIdx.x & 31;
  const int nsub = w >> 1;
  const int mh = w & 1;
  const int row0 = mblk * 64 + mh * 32;
  const int ntile = nblk * 4 + nsub;

  v8f acc0 = {}, acc1 = {};
#pragma unroll 4
  for (int kt = 0; kt < 16; ++kt) {    // K=512
    BFrag b = load_b_frag(W, ntile, kt, 16, lane);
    BFrag a0 = load_a_frag(A, INP, row0, kt * 32, lane);
    BFrag a1 = load_a_frag(A, INP, row0 + 16, kt * 32, lane);
    acc0 = wmma_bf16(a0.v, b.v, acc0);
    acc1 = wmma_bf16(a1.v, b.v, acc1);
  }
  const int colg = ntile * 16 + (lane & 15);
  const int half = lane >> 4;
  const float bv = bias[colg];
#pragma unroll
  for (int r = 0; r < 8; ++r) {
    out[(size_t)(row0 + half * 8 + r) * HID + colg] = (bf16_t)(acc0[r] + bv);
    out[(size_t)(row0 + 16 + half * 8 + r) * HID + colg] = (bf16_t)(acc1[r] + bv);
  }
}

// --------------------------- persistent LSTM -------------------------------

__device__ __forceinline__ void grid_barrier(unsigned* cnt, unsigned phase,
                                             unsigned nwg) {
  __threadfence();
  __syncthreads();
  if (threadIdx.x == 0) {
    __hip_atomic_fetch_add(cnt, 1u, __ATOMIC_RELEASE, __HIP_MEMORY_SCOPE_AGENT);
    while (__hip_atomic_load(cnt, __ATOMIC_ACQUIRE, __HIP_MEMORY_SCOPE_AGENT)
           < phase * nwg) {
      __builtin_amdgcn_s_sleep(2);
    }
  }
  __syncthreads();
}

__global__ void lstm_persistent_kernel(const bf16_t* __restrict__ xin_bf,
                                       const bf16_t* __restrict__ wxg_swz,
                                       const bf16_t* __restrict__ whg_swz,
                                       const float* __restrict__ bxg,
                                       const float* __restrict__ bhg,
                                       const float* __restrict__ h0,
                                       const float* __restrict__ c0,
                                       bf16_t* __restrict__ hbf,
                                       float* __restrict__ out,
                                       unsigned* __restrict__ cnt) {
  __shared__ bf16_t atile[2][BATCH][ACHUNK];  // 64 KB: TDM double buffer
  __shared__ float gatesLDS[4][64][16];       // 16 KB
  __shared__ float cLDS[NLAYER][64][16];      //  8 KB (cell state)

  const unsigned nwg = gridDim.x;             // 64
  const int cb = blockIdx.x * 16;             // hidden-column base of this WG
  const int w = threadIdx.x >> 5;
  const int lane = threadIdx.x & 31;
  const int q = w & 3;                        // gate: 0=i 1=f 2=g 3=o
  const int mh = w >> 2;                      // row half (32 rows each)
  const int half = lane >> 4;
  const int r16 = lane & 15;

  const unsigned lds_a0 = (unsigned)(unsigned long long)(void*)&atile[0][0][0];
  const unsigned lds_a1 = (unsigned)(unsigned long long)(void*)&atile[1][0][0];

  // ---- init: c-state into LDS, h0 -> bf16 global (this WG's columns) ----
  for (int e = threadIdx.x; e < 64 * 16; e += blockDim.x) {
    int row = e >> 4, col = e & 15;
#pragma unroll
    for (int l = 0; l < NLAYER; ++l) {
      cLDS[l][row][col] = c0[(size_t)(l * 64 + row) * HID + cb + col];
      hbf[(size_t)(l * 64 + row) * HID + cb + col] =
          (bf16_t)h0[(size_t)(l * 64 + row) * HID + cb + col];
    }
  }
  unsigned phase = 1;
  grid_barrier(cnt, phase, nwg);

  // ---- 512 steps x 2 layers ----
  for (int t = 0; t < SEQ; ++t) {
#pragma unroll 1
    for (int l = 0; l < NLAYER; ++l) {
      const bf16_t* A1 = (l == 0) ? (xin_bf + (size_t)t * BATCH * HID) : hbf;
      const bf16_t* A2 = hbf + (size_t)l * BATCH * HID;
      const bf16_t* Wx = wxg_swz + (size_t)l * HID * G4;
      const bf16_t* Wh = whg_swz + (size_t)l * HID * G4;
      const int ntile = q * 64 + blockIdx.x;   // of 256 gate n-tiles

      v8f acc0 = {}, acc1 = {};
#pragma unroll 1
      for (int pass = 0; pass < 2; ++pass) {
        const bf16_t* A = pass ? A2 : A1;
        const bf16_t* W = pass ? Wh : Wx;
        if (w == 0) tdm_load_a_tile(lds_a0, A);            // chunk 0 -> buf0
#pragma unroll 1
        for (int ch = 0; ch < HID / ACHUNK; ++ch) {        // 4 chunks
          if (w == 0) __builtin_amdgcn_s_wait_tensorcnt(0);
          __syncthreads();                                 // chunk ch ready
          if (w == 0 && ch + 1 < HID / ACHUNK)             // prefetch ch+1
            tdm_load_a_tile(((ch + 1) & 1) ? lds_a1 : lds_a0,
                            A + (ch + 1) * ACHUNK);
          const bf16_t* abuf = &atile[ch & 1][0][0];
#pragma unroll 4
          for (int kk = 0; kk < ACHUNK / 32; ++kk) {       // 8 k-tiles
            const int kt = ch * (ACHUNK / 32) + kk;
            BFrag b = load_b_frag(W, ntile, kt, 32, lane);
            BFrag a0, a1;
            {
              const bf16_t* p0 = abuf + (mh * 32 + r16) * ACHUNK
                                      + kk * 32 + half * 8;
              a0.h[0] = *(const v8bf*)(p0);
              a0.h[1] = *(const v8bf*)(p0 + 16);
              const bf16_t* p1 = p0 + 16 * ACHUNK;
              a1.h[0] = *(const v8bf*)(p1);
              a1.h[1] = *(const v8bf*)(p1 + 16);
            }
            acc0 = wmma_bf16(a0.v, b.v, acc0);
            acc1 = wmma_bf16(a1.v, b.v, acc1);
          }
          __syncthreads();                                 // buffer reuse
        }
      }

      // scatter gate tiles into LDS (C layout: vgpr r -> M = 8*half + r)
#pragma unroll
      for (int r = 0; r < 8; ++r) {
        gatesLDS[q][mh * 32 + half * 8 + r][r16] = acc0[r];
        gatesLDS[q][mh * 32 + 16 + half * 8 + r][r16] = acc1[r];
      }
      __syncthreads();

      // LSTM cell update for this WG's 64x16 slice
      for (int e = threadIdx.x; e < 64 * 16; e += blockDim.x) {
        int row = e >> 4, col = e & 15;
        int cg = cb + col;
        const float* bx = bxg + (size_t)l * G4;
        const float* bh = bhg + (size_t)l * G4;
        float iv = gatesLDS[0][row][col] + bx[cg] + bh[cg];
        float fv = gatesLDS[1][row][col] + bx[HID + cg] + bh[HID + cg];
        float gv = gatesLDS[2][row][col] + bx[2 * HID + cg] + bh[2 * HID + cg];
        float ov = gatesLDS[3][row][col] + bx[3 * HID + cg] + bh[3 * HID + cg];
        float c1 = sigm(fv) * cLDS[l][row][col] + sigm(iv) * tanhf(gv);
        float h1 = sigm(ov) * c1;
        cLDS[l][row][col] = c1;
        hbf[(size_t)(l * 64 + row) * HID + cg] = (bf16_t)h1;
        if (l == 1)
          out[((size_t)t * 64 + row) * HID + cg] = h1;
      }
      grid_barrier(cnt, ++phase, nwg);
    }
  }
}

// --------------------------- launch ----------------------------------------

extern "C" void kernel_launch(void* const* d_in, const int* in_sizes, int n_in,
                              void* d_out, int out_size, void* d_ws, size_t ws_size,
                              hipStream_t stream) {
  (void)in_sizes; (void)n_in; (void)out_size; (void)ws_size;
  const float* x   = (const float*)d_in[0];  // [512,64,512]
  const float* Wi  = (const float*)d_in[1];  // [512,1024]
  const float* bi  = (const float*)d_in[2];  // [1024]
  const float* Wxg = (const float*)d_in[3];  // [2,1024,4096]
  const float* Whg = (const float*)d_in[4];  // [2,1024,4096]
  const float* bxg = (const float*)d_in[5];  // [2,4096]
  const float* bhg = (const float*)d_in[6];  // [2,4096]
  const float* h0  = (const float*)d_in[7];  // [2,64,1024]
  const float* c0  = (const float*)d_in[8];  // [2,64,1024]
  float* out = (float*)d_out;                // [512,64,1024]

  char* ws = (char*)d_ws;
  size_t o = 0;
  unsigned* cnt = (unsigned*)(ws + o);                 o += 256;
  bf16_t* x_bf    = (bf16_t*)(ws + o);                 o += (size_t)SEQ * BATCH * INP * 2;
  bf16_t* wi_swz  = (bf16_t*)(ws + o);                 o += (size_t)INP * HID * 2;
  bf16_t* wxg_swz = (bf16_t*)(ws + o);                 o += (size_t)NLAYER * HID * G4 * 2;
  bf16_t* whg_swz = (bf16_t*)(ws + o);                 o += (size_t)NLAYER * HID * G4 * 2;
  bf16_t* xin_bf  = (bf16_t*)(ws + o);                 o += (size_t)SEQ * BATCH * HID * 2;
  bf16_t* hbf     = (bf16_t*)(ws + o);                 o += (size_t)NLAYER * BATCH * HID * 2;

  hipMemsetAsync(cnt, 0, 256, stream);

  cvt_f32_bf16_kernel<<<4096, 256, 0, stream>>>(x, x_bf, SEQ * BATCH * INP);
  swizzle_w_kernel<<<1024, 256, 0, stream>>>(Wi, wi_swz, INP, HID);
  swizzle_w_kernel<<<2048, 256, 0, stream>>>(Wxg, wxg_swz, HID, G4);
  swizzle_w_kernel<<<2048, 256, 0, stream>>>(Wxg + (size_t)HID * G4,
                                             wxg_swz + (size_t)HID * G4, HID, G4);
  swizzle_w_kernel<<<2048, 256, 0, stream>>>(Whg, whg_swz, HID, G4);
  swizzle_w_kernel<<<2048, 256, 0, stream>>>(Whg + (size_t)HID * G4,
                                             whg_swz + (size_t)HID * G4, HID, G4);

  gemm_xin_kernel<<<8192, 256, 0, stream>>>(x_bf, wi_swz, bi, xin_bf);

  lstm_persistent_kernel<<<64, 256, 0, stream>>>(xin_bf, wxg_swz, whg_swz,
                                                 bxg, bhg, h0, c0, hbf, out, cnt);
}
